// DotProductAttention_19138374271217
// MI455X (gfx1250) — matrix operand order repaired and checked by reading, not compiled, not run on hardware
//
#include <hip/hip_runtime.h>

// CDNA5 / gfx1250 fused dot-product attention:
//   score = Q @ V^T (no 1/sqrt(d) scaling), attn = softmax(score), ctx = attn @ V
//
// All fp32->fp16 hi/lo error-compensated splits (x ~= hi+lo, GEMMs compute
// hi*hi + hi*lo + lo*hi with V_WMMA_F32_16X16X32_F16, ~22 mantissa bits) are
// hoisted OUT of the GEMM hot loops:
//   * prep kernels split Q and V once into fp16 pairs in d_ws (V also in a
//     transposed copy for the context GEMM),
//   * softmax (phase 2) emits P directly as fp16 hi/lo into LDS for phase 3.
// Hot loops are pure b128 global/LDS loads -> WMMA.

typedef __attribute__((ext_vector_type(16))) _Float16 v16h;
typedef __attribute__((ext_vector_type(8)))  _Float16 v8h;
typedef __attribute__((ext_vector_type(4)))  _Float16 v4h;
typedef __attribute__((ext_vector_type(8)))  float    v8f;

#define BATCH 16
#define LSEQ  2048
#define DDIM  1024
#define SSTR  2056                        // fp32 score row stride (floats)
#define PSTR  2064                        // fp16 P row stride (halfs, 32B pad)
#define SMEM_BYTES (16 * SSTR * 4 + 2 * 16 * PSTR * 2)   // 263,680 B
#define NELEM ((size_t)BATCH * LSEQ * DDIM)

#define WMMA_F16(A, B, C) \
    __builtin_amdgcn_wmma_f32_16x16x32_f16(false, (A), false, (B), (short)0, (C), false, false)

#define CAT16(H0, H1) \
    __builtin_shufflevector((H0), (H1), 0, 1, 2, 3, 4, 5, 6, 7, 8, 9, 10, 11, 12, 13, 14, 15)

// ---------------------------------------------------------------------------
// Prep A: Q(fp32) -> Qhi/Qlo fp16, same row-major layout. float4 per thread.
// ---------------------------------------------------------------------------
__global__ __launch_bounds__(256)
void split_q_kernel(const float* __restrict__ Q,
                    _Float16* __restrict__ Qhi, _Float16* __restrict__ Qlo)
{
    const size_t i = ((size_t)blockIdx.x * 256 + threadIdx.x) * 4;
    const float4 x = *(const float4*)(Q + i);
    v4h h, l;
    h[0] = (_Float16)x.x; l[0] = (_Float16)(x.x - (float)h[0]);
    h[1] = (_Float16)x.y; l[1] = (_Float16)(x.y - (float)h[1]);
    h[2] = (_Float16)x.z; l[2] = (_Float16)(x.z - (float)h[2]);
    h[3] = (_Float16)x.w; l[3] = (_Float16)(x.w - (float)h[3]);
    *(v4h*)(Qhi + i) = h;
    *(v4h*)(Qlo + i) = l;
}

// ---------------------------------------------------------------------------
// Prep B: V(fp32, (B,Lk,D)) -> Vhi/Vlo (fp16, same layout) + Vthi/Vtlo (fp16,
// transposed (B,D,Lk)). 32x32 tile transpose through LDS, coalesced all ways.
// ---------------------------------------------------------------------------
__global__ __launch_bounds__(256)
void split_v_kernel(const float* __restrict__ V,
                    _Float16* __restrict__ Vhi,  _Float16* __restrict__ Vlo,
                    _Float16* __restrict__ Vthi, _Float16* __restrict__ Vtlo)
{
    __shared__ _Float16 th[32][33];
    __shared__ _Float16 tl[32][33];

    const int b  = blockIdx.z;
    const int k0 = blockIdx.y << 5;
    const int d0 = blockIdx.x << 5;
    const int tx = threadIdx.x;
    const int ty = threadIdx.y;

#pragma unroll
    for (int i = 0; i < 4; ++i) {
        const int kk = ty + 8 * i;
        const size_t idx = ((size_t)b * LSEQ + k0 + kk) * DDIM + d0 + tx;
        float    x = V[idx];
        _Float16 h = (_Float16)x;
        _Float16 l = (_Float16)(x - (float)h);
        Vhi[idx] = h;
        Vlo[idx] = l;
        th[kk][tx] = h;
        tl[kk][tx] = l;
    }
    __syncthreads();
#pragma unroll
    for (int i = 0; i < 4; ++i) {
        const int dd = ty + 8 * i;
        const size_t idx = ((size_t)b * DDIM + d0 + dd) * LSEQ + k0 + tx;
        Vthi[idx] = th[tx][dd];
        Vtlo[idx] = tl[tx][dd];
    }
}

// ---------------------------------------------------------------------------
// Fused attention: one 16-query tile per workgroup, 16 waves (512 threads)
// ---------------------------------------------------------------------------
__global__ __launch_bounds__(512)
void attn_fused_kernel(const _Float16* __restrict__ Qhi,
                       const _Float16* __restrict__ Qlo,
                       const _Float16* __restrict__ Vhi,
                       const _Float16* __restrict__ Vlo,
                       const _Float16* __restrict__ Vthi,
                       const _Float16* __restrict__ Vtlo,
                       float* __restrict__ ctx_out,
                       float* __restrict__ attn_out)
{
    extern __shared__ float S[];                 // [16][SSTR] fp32 scores
    _Float16* PH = (_Float16*)(S + 16 * SSTR);   // [16][PSTR] fp16 P hi
    _Float16* PL = PH + 16 * PSTR;               // [16][PSTR] fp16 P lo

    const int tid   = threadIdx.x;
    const int lane  = tid & 31;
    const int wave  = tid >> 5;          // 16 waves
    const int row16 = lane & 15;         // M (or N) index within a 16-wide tile
    const int half  = lane >> 4;         // lane-half selector

    const int b     = blockIdx.x >> 7;          // 128 q-tiles per batch
    const int qbase = (blockIdx.x & 127) << 4;  // 16 query rows per block

    const _Float16* Qhb  = Qhi  + ((size_t)b * LSEQ + qbase) * DDIM;
    const _Float16* Qlb  = Qlo  + ((size_t)b * LSEQ + qbase) * DDIM;
    const _Float16* Vhb  = Vhi  + (size_t)b * LSEQ * DDIM;
    const _Float16* Vlb  = Vlo  + (size_t)b * LSEQ * DDIM;
    const _Float16* Vthb = Vthi + (size_t)b * DDIM * LSEQ;
    const _Float16* Vtlb = Vtlo + (size_t)b * DDIM * LSEQ;

    // ---------------- Phase 1: S(16x2048) = Qtile @ V^T  (fp16x3 WMMA) ----
    v8f acc[8] = {};
    const int ncol0 = wave << 7;         // 128 key columns per wave

    for (int d0 = 0; d0 < DDIM; d0 += 32) {
        // A fragment: row M = row16, one contiguous K-run {16h..16h+15}
        const size_t qoff = (size_t)row16 * DDIM + d0 + 16 * half;
        const v16h a_hi = CAT16(*(const v8h*)(Qhb + qoff),
                                *(const v8h*)(Qhb + qoff + 8));
        const v16h a_lo = CAT16(*(const v8h*)(Qlb + qoff),
                                *(const v8h*)(Qlb + qoff + 8));
#pragma unroll
        for (int nt = 0; nt < 8; ++nt) {
            // B fragment: N = key row, element i <-> K = i + 16h (contiguous d)
            const size_t off = (size_t)(ncol0 + nt * 16 + row16) * DDIM
                             + d0 + 16 * half;
            const v16h b_hi = *(const v16h*)(Vhb + off);
            const v16h b_lo = *(const v16h*)(Vlb + off);
            __builtin_prefetch(Vhb + off + 32, 0, 1);
            acc[nt] = WMMA_F16(a_hi, b_hi, acc[nt]);
            acc[nt] = WMMA_F16(a_hi, b_lo, acc[nt]);
            acc[nt] = WMMA_F16(a_lo, b_hi, acc[nt]);
        }
    }

    // spill score tiles to LDS: C/D layout => M = r + 8*half, N = row16
#pragma unroll
    for (int nt = 0; nt < 8; ++nt) {
#pragma unroll
        for (int r = 0; r < 8; ++r) {
            S[(size_t)(r + 8 * half) * SSTR + ncol0 + nt * 16 + row16] = acc[nt][r];
        }
    }
    __syncthreads();

    // ---------------- Phase 2: softmax — wave w owns score row w ----------
    {
        float*    Sr  = S  + (size_t)wave * SSTR;
        _Float16* PHr = PH + (size_t)wave * PSTR;
        _Float16* PLr = PL + (size_t)wave * PSTR;

        float m = -3.0e38f;
        for (int j = lane; j < LSEQ; j += 32) m = fmaxf(m, Sr[j]);
#pragma unroll
        for (int off = 16; off > 0; off >>= 1) m = fmaxf(m, __shfl_xor(m, off, 32));

        float sum = 0.0f;
        for (int j = lane; j < LSEQ; j += 32) {
            float p = __expf(Sr[j] - m);
            Sr[j]   = p;
            sum    += p;
        }
#pragma unroll
        for (int off = 16; off > 0; off >>= 1) sum += __shfl_xor(sum, off, 32);

        const float inv  = 1.0f / sum;
        float*      arow = attn_out + ((size_t)b * LSEQ + qbase + wave) * LSEQ;
        for (int j = lane; j < LSEQ; j += 32) {
            float    p = Sr[j] * inv;
            _Float16 h = (_Float16)p;
            PHr[j] = h;
            PLr[j] = (_Float16)(p - (float)h);   // fp16 hi/lo P for phase 3
            arow[j] = p;                         // coalesced attn output
        }
    }
    __syncthreads();

    // ---------------- Phase 3: ctx(16x1024) = P(16x2048) @ V(2048x1024) ---
    v8f cacc[4] = {};
    const int cn0 = wave << 6;           // 64 output columns per wave

    for (int k0 = 0; k0 < LSEQ; k0 += 32) {
        // A fragment from LDS fp16 P hi/lo: one contiguous 16-element K-run {16h..}
        const size_t poff = (size_t)row16 * PSTR + k0 + 16 * half;
        const v16h a_hi = CAT16(*(const v8h*)(PH + poff),
                                *(const v8h*)(PH + poff + 8));
        const v16h a_lo = CAT16(*(const v8h*)(PL + poff),
                                *(const v8h*)(PL + poff + 8));
#pragma unroll
        for (int nt = 0; nt < 4; ++nt) {
            // B fragment from transposed V: N = d column, K = k contiguous
            const size_t off = (size_t)(cn0 + nt * 16 + row16) * LSEQ
                             + k0 + 16 * half;
            const v16h b_hi = *(const v16h*)(Vthb + off);
            const v16h b_lo = *(const v16h*)(Vtlb + off);
            __builtin_prefetch(Vthb + off + 32, 0, 1);
            cacc[nt] = WMMA_F16(a_hi, b_hi, cacc[nt]);
            cacc[nt] = WMMA_F16(a_hi, b_lo, cacc[nt]);
            cacc[nt] = WMMA_F16(a_lo, b_hi, cacc[nt]);
        }
    }

    // store context tile: M = r + 8*half, N = cn0 + nt*16 + row16
    float* crow = ctx_out + ((size_t)b * LSEQ + qbase) * DDIM;
#pragma unroll
    for (int nt = 0; nt < 4; ++nt) {
#pragma unroll
        for (int r = 0; r < 8; ++r) {
            crow[(size_t)(r + 8 * half) * DDIM + cn0 + nt * 16 + row16] = cacc[nt][r];
        }
    }
}

extern "C" void kernel_launch(void* const* d_in, const int* in_sizes, int n_in,
                              void* d_out, int out_size, void* d_ws, size_t ws_size,
                              hipStream_t stream) {
    (void)in_sizes; (void)n_in; (void)out_size; (void)ws_size;

    const float* Q = (const float*)d_in[0];
    const float* V = (const float*)d_in[1];
    float* ctx  = (float*)d_out;                  // (B, Lq, D)
    float* attn = ctx + NELEM;                    // (B, Lq, Lk)

    // workspace: 6 fp16 arrays of B*L*D elements = 384 MiB total
    _Float16* Qhi  = (_Float16*)d_ws;
    _Float16* Qlo  = Qhi  + NELEM;
    _Float16* Vhi  = Qlo  + NELEM;
    _Float16* Vlo  = Vhi  + NELEM;
    _Float16* Vthi = Vlo  + NELEM;
    _Float16* Vtlo = Vthi + NELEM;

    split_q_kernel<<<dim3((unsigned)(NELEM / (256 * 4))), 256, 0, stream>>>(Q, Qhi, Qlo);
    split_v_kernel<<<dim3(DDIM / 32, LSEQ / 32, BATCH), dim3(32, 8), 0, stream>>>(
        V, Vhi, Vlo, Vthi, Vtlo);

    hipFuncSetAttribute(reinterpret_cast<const void*>(attn_fused_kernel),
                        hipFuncAttributeMaxDynamicSharedMemorySize, SMEM_BYTES);
    attn_fused_kernel<<<dim3(BATCH * (LSEQ / 16)), 512, SMEM_BYTES, stream>>>(
        Qhi, Qlo, Vhi, Vlo, Vthi, Vtlo, ctx, attn);
}
